// VAE_decoder_85925115724585
// MI455X (gfx1250) — matrix operand order repaired and checked by reading, not compiled, not run on hardware
//
#include <hip/hip_runtime.h>
#include <stdint.h>

// ---------------------------------------------------------------------------
// VAE decoder, fused bf16-WMMA pipeline for gfx1250 (MI455X).
//   stage1: h1[e]   = elu(x @ w1[e] + b1[e])            e=0..7
//   stage2: moe     = sum_e para[e]*elu(h1[e]@w2[e]+b2) (expert loop in-kernel)
//   stage3: s4      = elu(concat(moe,z) @ w3 + b3)
//   stage4: s5      = elu(concat(s4 ,z) @ w4 + b4)
//   stage5: out     = elu(concat(s5 ,z) @ w5 + b5)  (fp32)
// concat(.,z) is free: z columns are pre-placed once in each stage buffer.
//
// Block tile 256x64 (8 waves; each wave owns TWO 16-row strips so every
// B fragment feeds two WMMAs), double-buffered LDS tiles fed by
// GLOBAL_LOAD_ASYNC_TO_LDS (ASYNCcnt DMA, no VGPR round-trip), prefetch
// distance 1, one workgroup barrier per K-step.
// ---------------------------------------------------------------------------

typedef __attribute__((ext_vector_type(16))) __bf16 v16bf;
typedef __attribute__((ext_vector_type(8)))  float  v8f;

constexpr int BATCH = 32768;

union Frag {
    uint4 u4[2];
    v16bf v;
};

__device__ __forceinline__ float elu1(float x) {
    return x > 0.f ? x : (__expf(x) - 1.f);
}

// Low 32 bits of a generic pointer to __shared__ = LDS byte offset
// (flat aperture lives in the upper 32 bits on CDNA5).
__device__ __forceinline__ unsigned lds_off32(const void* p) {
    return (unsigned)(uintptr_t)p;
}

// Async DMA: 16 bytes global -> LDS, per lane; tracked with ASYNCcnt.
__device__ __forceinline__ void async_copy_b128(unsigned lds, const void* gaddr) {
    asm volatile("global_load_async_to_lds_b128 %0, %1, off"
                 :: "v"(lds), "v"(gaddr)
                 : "memory");
}

__device__ __forceinline__ void wait_async_zero() {
    asm volatile("s_wait_asynccnt 0x0" ::: "memory");
}

// ---------------------------------------------------------------------------
// Prep: bf16 convert + build concat inputs.
//   bufx[b, 0:512]=motions, bufx[b,512:576]=z
//   s3in[b,576:640]=z ; s4in[b,256:320]=z ; s5in[b,576:640]=z
// ---------------------------------------------------------------------------
__global__ void vae_prep_inputs(const float* __restrict__ motions,
                                const float* __restrict__ z,
                                __bf16* __restrict__ bufx,
                                __bf16* __restrict__ s3in,
                                __bf16* __restrict__ s4in,
                                __bf16* __restrict__ s5in) {
    int idx = blockIdx.x * blockDim.x + threadIdx.x;
    if (idx >= BATCH * 576) return;
    int b = idx / 576, c = idx - b * 576;
    float v;
    if (c < 512) {
        v = motions[(size_t)b * 512 + c];
    } else {
        int j = c - 512;
        v = z[(size_t)b * 64 + j];
        __bf16 bv = (__bf16)v;
        s3in[(size_t)b * 640 + 576 + j] = bv;
        s4in[(size_t)b * 320 + 256 + j] = bv;
        s5in[(size_t)b * 640 + 576 + j] = bv;
    }
    bufx[(size_t)b * 576 + c] = (__bf16)v;
}

// Convert W[e][k][n] (fp32, row-major KxN) -> Wt[e][n][k] (bf16, N-major).
__global__ void vae_conv_transpose(const float* __restrict__ src,
                                   __bf16* __restrict__ dst,
                                   int K, int N, int total) {
    int idx = blockIdx.x * blockDim.x + threadIdx.x;
    if (idx >= total) return;
    int kn = K * N;
    int e = idx / kn;
    int r = idx - e * kn;
    int k = r / N;
    int n = r - k * N;
    dst[(size_t)e * kn + (size_t)n * K + k] = (__bf16)src[idx];
}

// ---------------------------------------------------------------------------
// Generic bf16 WMMA GEMM: C[m,n] = act(A[m, :K] @ Wt[n, :K]^T + bias[n])
// Block tile 256x64, 256 threads = 8 waves; wave w owns rows
// [w*32, w*32+32) as two 16-row strips, and four 16-wide n-tiles
// (B fragments reused across both strips: 8 WMMA per 12 b128 LDS reads).
// LDS rows padded to 40 bf16 (20 banks) so b128 fragment reads are
// bank-conflict free.
// Fragment lane maps (per CDNA5 ISA 7.12.2):
//   A 16x32 bf16 : row = lane&15; elems 0..7 -> k = (lane>>4)*8 + j
//                                 elems 8..15 -> k = 16 + (lane>>4)*8 + (j-8)
//   B 32x16 bf16 : col = lane&15; elems j -> same k as A elem j
//   C 16x16 f32  : col = lane&15; elem r -> row = (lane>>4)*8 + r
// ---------------------------------------------------------------------------
__global__ __launch_bounds__(256, 2) void vae_gemm_bf16_wmma(
    const __bf16* __restrict__ A, long long strideAe, int lda,
    const __bf16* __restrict__ Wt, long long strideWe,       // [N][K] bf16
    const float*  __restrict__ bias, int strideBe,
    int K, int N,
    __bf16* __restrict__ Cbf, long long strideCe, int ldc, int ccol,
    float*  __restrict__ Cf, int ldcf, int doElu) {
    __shared__ alignas(16) __bf16 Alds[2][256 * 40];
    __shared__ alignas(16) __bf16 Blds[2][64 * 40];

    const int e = blockIdx.z;
    A    += (long long)e * strideAe;
    Wt   += (long long)e * strideWe;
    bias += (long long)e * strideBe;

    const int m0   = blockIdx.y * 256;
    const int n0   = blockIdx.x * 64;
    const int tid  = threadIdx.x;
    const int lane = tid & 31;
    const int wave = tid >> 5;   // 0..7 -> 32-row strip pair
    const int lhi  = lane >> 4;  // 0/1
    const int lrow = lane & 15;

    // Async staging: A tile 256x32 = 1024 b128 chunks (4/thread), B 1/thread.
    const int arb = tid >> 2;                 // A base row; +64 per chunk
    const int ac  = (tid & 3) << 3;           // A col within 32
    const int br  = tid >> 2;                 // B n-row
    const int bc  = (tid & 3) << 3;

    auto issue_tile = [&](int k0, int buf) {
        #pragma unroll
        for (int it = 0; it < 4; ++it) {
            int r = arb + it * 64;
            async_copy_b128(lds_off32(&Alds[buf][r * 40 + ac]),
                            A + (long long)(m0 + r) * lda + k0 + ac);
        }
        async_copy_b128(lds_off32(&Blds[buf][br * 40 + bc]),
                        Wt + (long long)(n0 + br) * K + k0 + bc);
    };

    v8f acc[2][4];
    #pragma unroll
    for (int s = 0; s < 2; ++s)
        #pragma unroll
        for (int t = 0; t < 4; ++t)
            #pragma unroll
            for (int r = 0; r < 8; ++r) acc[s][t][r] = 0.f;

    const int nk = K >> 5;
    issue_tile(0, 0);
    for (int ki = 0; ki < nk; ++ki) {
        const int buf = ki & 1;
        wait_async_zero();      // tile ki resident (tile ki+1 not yet issued)
        __syncthreads();        // all waves' DMA done; prior readers of buf^1 past
        if (ki + 1 < nk) issue_tile((ki + 1) << 5, buf ^ 1);

        Frag af[2];
        #pragma unroll
        for (int s = 0; s < 2; ++s) {
            const int row = wave * 32 + s * 16 + lrow;
            af[s].u4[0] = *reinterpret_cast<const uint4*>(
                &Alds[buf][row * 40 + lhi * 8]);
            af[s].u4[1] = *reinterpret_cast<const uint4*>(
                &Alds[buf][row * 40 + 16 + lhi * 8]);
        }
        #pragma unroll
        for (int t = 0; t < 4; ++t) {
            Frag bf;
            bf.u4[0] = *reinterpret_cast<const uint4*>(
                &Blds[buf][(t * 16 + lrow) * 40 + lhi * 8]);
            bf.u4[1] = *reinterpret_cast<const uint4*>(
                &Blds[buf][(t * 16 + lrow) * 40 + 16 + lhi * 8]);
            acc[0][t] = __builtin_amdgcn_wmma_f32_16x16x32_bf16(
                false, af[0].v, false, bf.v, (short)0, acc[0][t], false, false);
            acc[1][t] = __builtin_amdgcn_wmma_f32_16x16x32_bf16(
                false, af[1].v, false, bf.v, (short)0, acc[1][t], false, false);
        }
    }

    #pragma unroll
    for (int s = 0; s < 2; ++s)
        #pragma unroll
        for (int t = 0; t < 4; ++t) {
            int n = n0 + t * 16 + lrow;
            float bv = bias[n];
            #pragma unroll
            for (int r = 0; r < 8; ++r) {
                int m = m0 + wave * 32 + s * 16 + lhi * 8 + r;
                float v = acc[s][t][r] + bv;
                if (doElu) v = elu1(v);
                if (Cbf)
                    Cbf[(long long)blockIdx.z * strideCe + (long long)m * ldc + ccol + n] =
                        (__bf16)v;
                if (Cf) Cf[(long long)m * ldcf + n] = v;
            }
        }
}

// ---------------------------------------------------------------------------
// MoE stage 2 with in-register expert sum:
//   C[m,n] = sum_e para[e] * elu(H1[e,m,:256] @ W2t[e][n][:256]^T + B2[e,n])
// 8 experts x 8 K-steps flattened into one 64-step async pipeline.
// ---------------------------------------------------------------------------
__global__ __launch_bounds__(256, 2) void vae_moe_stage2_wmma(
    const __bf16* __restrict__ H1,   // [8][BATCH][256]
    const __bf16* __restrict__ W2t,  // [8][576][256]
    const float*  __restrict__ B2,   // [8][576]
    const float*  __restrict__ para, // [8]
    __bf16* __restrict__ Cbf, int ldc) {
    __shared__ alignas(16) __bf16 Alds[2][256 * 40];
    __shared__ alignas(16) __bf16 Blds[2][64 * 40];

    const int m0   = blockIdx.y * 256;
    const int n0   = blockIdx.x * 64;
    const int tid  = threadIdx.x;
    const int lane = tid & 31;
    const int wave = tid >> 5;
    const int lhi  = lane >> 4;
    const int lrow = lane & 15;

    const int arb = tid >> 2;
    const int ac  = (tid & 3) << 3;
    const int br  = tid >> 2;
    const int bc  = (tid & 3) << 3;

    auto issue_step = [&](int st, int buf) {
        const int e  = st >> 3;
        const int k0 = (st & 7) << 5;
        const __bf16* Ae  = H1  + (long long)e * BATCH * 256;
        const __bf16* Wte = W2t + (long long)e * 576 * 256;
        #pragma unroll
        for (int it = 0; it < 4; ++it) {
            int r = arb + it * 64;
            async_copy_b128(lds_off32(&Alds[buf][r * 40 + ac]),
                            Ae + (long long)(m0 + r) * 256 + k0 + ac);
        }
        async_copy_b128(lds_off32(&Blds[buf][br * 40 + bc]),
                        Wte + (long long)(n0 + br) * 256 + k0 + bc);
    };

    float sum[2][4][8];
    #pragma unroll
    for (int s = 0; s < 2; ++s)
        #pragma unroll
        for (int t = 0; t < 4; ++t)
            #pragma unroll
            for (int r = 0; r < 8; ++r) sum[s][t][r] = 0.f;

    v8f acc[2][4];
    issue_step(0, 0);
    for (int st = 0; st < 64; ++st) {
        const int buf = st & 1;
        if ((st & 7) == 0) {
            #pragma unroll
            for (int s = 0; s < 2; ++s)
                #pragma unroll
                for (int t = 0; t < 4; ++t)
                    #pragma unroll
                    for (int r = 0; r < 8; ++r) acc[s][t][r] = 0.f;
        }
        wait_async_zero();
        __syncthreads();
        if (st + 1 < 64) issue_step(st + 1, buf ^ 1);

        Frag af[2];
        #pragma unroll
        for (int s = 0; s < 2; ++s) {
            const int row = wave * 32 + s * 16 + lrow;
            af[s].u4[0] = *reinterpret_cast<const uint4*>(
                &Alds[buf][row * 40 + lhi * 8]);
            af[s].u4[1] = *reinterpret_cast<const uint4*>(
                &Alds[buf][row * 40 + 16 + lhi * 8]);
        }
        #pragma unroll
        for (int t = 0; t < 4; ++t) {
            Frag bf;
            bf.u4[0] = *reinterpret_cast<const uint4*>(
                &Blds[buf][(t * 16 + lrow) * 40 + lhi * 8]);
            bf.u4[1] = *reinterpret_cast<const uint4*>(
                &Blds[buf][(t * 16 + lrow) * 40 + 16 + lhi * 8]);
            acc[0][t] = __builtin_amdgcn_wmma_f32_16x16x32_bf16(
                false, af[0].v, false, bf.v, (short)0, acc[0][t], false, false);
            acc[1][t] = __builtin_amdgcn_wmma_f32_16x16x32_bf16(
                false, af[1].v, false, bf.v, (short)0, acc[1][t], false, false);
        }

        if ((st & 7) == 7) {
            const int e = st >> 3;
            float pe = para[e];
            #pragma unroll
            for (int s = 0; s < 2; ++s)
                #pragma unroll
                for (int t = 0; t < 4; ++t) {
                    float bv = B2[e * 576 + n0 + t * 16 + lrow];
                    #pragma unroll
                    for (int r = 0; r < 8; ++r)
                        sum[s][t][r] += pe * elu1(acc[s][t][r] + bv);
                }
        }
    }

    // Un-normalized expert sum (matches reference), written as bf16 into s3in.
    #pragma unroll
    for (int s = 0; s < 2; ++s)
        #pragma unroll
        for (int t = 0; t < 4; ++t) {
            int n = n0 + t * 16 + lrow;
            #pragma unroll
            for (int r = 0; r < 8; ++r) {
                int m = m0 + wave * 32 + s * 16 + lhi * 8 + r;
                Cbf[(long long)m * ldc + n] = (__bf16)sum[s][t][r];
            }
        }
}

// ---------------------------------------------------------------------------
extern "C" void kernel_launch(void* const* d_in, const int* in_sizes, int n_in,
                              void* d_out, int out_size, void* d_ws, size_t ws_size,
                              hipStream_t stream) {
    (void)in_sizes; (void)n_in; (void)out_size; (void)ws_size;
    const float* motions = (const float*)d_in[0];
    const float* z    = (const float*)d_in[1];
    const float* w1   = (const float*)d_in[2];
    const float* b1   = (const float*)d_in[3];
    const float* w2   = (const float*)d_in[4];
    const float* b2   = (const float*)d_in[5];
    const float* para = (const float*)d_in[6];
    const float* w3   = (const float*)d_in[7];
    const float* b3   = (const float*)d_in[8];
    const float* w4   = (const float*)d_in[9];
    const float* b4   = (const float*)d_in[10];
    const float* w5   = (const float*)d_in[11];
    const float* b5   = (const float*)d_in[12];
    float* out = (float*)d_out;

    char* ws = (char*)d_ws;
    size_t off = 0;
    auto alloc = [&](size_t bytes) {
        char* p = ws + off;
        off += (bytes + 255) & ~(size_t)255;
        return p;
    };
    __bf16* bufx = (__bf16*)alloc((size_t)BATCH * 576 * 2);
    __bf16* h1   = (__bf16*)alloc((size_t)8 * BATCH * 256 * 2);
    __bf16* s3in = (__bf16*)alloc((size_t)BATCH * 640 * 2);
    __bf16* s4in = (__bf16*)alloc((size_t)BATCH * 320 * 2);
    __bf16* s5in = (__bf16*)alloc((size_t)BATCH * 640 * 2);
    __bf16* w1t  = (__bf16*)alloc((size_t)8 * 576 * 256 * 2);
    __bf16* w2t  = (__bf16*)alloc((size_t)8 * 256 * 576 * 2);
    __bf16* w3t  = (__bf16*)alloc((size_t)640 * 256 * 2);
    __bf16* w4t  = (__bf16*)alloc((size_t)320 * 576 * 2);
    __bf16* w5t  = (__bf16*)alloc((size_t)640 * 128 * 2);

    // --- prep ---
    {
        int total = BATCH * 576;
        vae_prep_inputs<<<(total + 255) / 256, 256, 0, stream>>>(
            motions, z, bufx, s3in, s4in, s5in);
    }
    auto convT = [&](const float* src, __bf16* dst, int K, int N, int E) {
        int total = E * K * N;
        vae_conv_transpose<<<(total + 255) / 256, 256, 0, stream>>>(
            src, dst, K, N, total);
    };
    convT(w1, w1t, 576, 256, 8);
    convT(w2, w2t, 256, 576, 8);
    convT(w3, w3t, 640, 256, 1);
    convT(w4, w4t, 320, 576, 1);
    convT(w5, w5t, 640, 128, 1);

    // --- stage 1: per-expert h1 = elu(x @ w1[e] + b1[e]) ---
    vae_gemm_bf16_wmma<<<dim3(256 / 64, BATCH / 256, 8), 256, 0, stream>>>(
        bufx, 0LL, 576,
        w1t, (long long)256 * 576,
        b1, 256,
        576, 256,
        h1, (long long)BATCH * 256, 256, 0,
        nullptr, 0, 1);

    // --- stage 2: expert-sum into s3in[:, 0:576] ---
    vae_moe_stage2_wmma<<<dim3(576 / 64, BATCH / 256), 256, 0, stream>>>(
        h1, w2t, b2, para, s3in, 640);

    // --- stage 3: s4in[:, 0:256] = elu(s3in @ w3 + b3) ---
    vae_gemm_bf16_wmma<<<dim3(256 / 64, BATCH / 256, 1), 256, 0, stream>>>(
        s3in, 0LL, 640,
        w3t, 0LL,
        b3, 0,
        640, 256,
        s4in, 0LL, 320, 0,
        nullptr, 0, 1);

    // --- stage 4: s5in[:, 0:576] = elu(s4in @ w4 + b4) ---
    vae_gemm_bf16_wmma<<<dim3(576 / 64, BATCH / 256, 1), 256, 0, stream>>>(
        s4in, 0LL, 320,
        w4t, 0LL,
        b4, 0,
        320, 576,
        s5in, 0LL, 640, 0,
        nullptr, 0, 1);

    // --- stage 5: out = elu(s5in @ w5 + b5) (fp32) ---
    vae_gemm_bf16_wmma<<<dim3(128 / 64, BATCH / 256, 1), 256, 0, stream>>>(
        s5in, 0LL, 640,
        w5t, 0LL,
        b5, 0,
        640, 128,
        nullptr, 0LL, 0, 0,
        out, 128, 1);
}
